// CausalSelfAttention_54623394071323
// MI455X (gfx1250) — compile-verified, matrix operand order checked by reading, not run on hardware
//
#include <hip/hip_runtime.h>
#include <hip/hip_bf16.h>

// Causal self-attention for MI455X (gfx1250, wave32, WMMA).
// B=4, T=2048, D=1024, H=16, HD=64.  All GEMMs in bf16 via
// v_wmma_f32_16x16x32_bf16 (f32 accumulate), flash-style attention.
// GEMM weight tiles now stream into LDS with the CDNA5 async path
// (global_load_async_to_lds_b128 + ASYNCcnt) and double buffering.

#define BB   4
#define TT   2048
#define DD   1024
#define HH   16
#define HDD  64
#define MTOT (BB * TT)   // 8192

typedef __bf16 bf16;
typedef __attribute__((ext_vector_type(4)))  __bf16 bf16x4;
typedef __attribute__((ext_vector_type(8)))  __bf16 bf16x8;
typedef __attribute__((ext_vector_type(16))) __bf16 bf16x16;
typedef __attribute__((ext_vector_type(8)))  float  f32x8;

__device__ __forceinline__ f32x8 zero_f32x8() {
  f32x8 z;
#pragma unroll
  for (int i = 0; i < 8; ++i) z[i] = 0.0f;
  return z;
}

__device__ __forceinline__ f32x8 wmma_bf16(bf16x16 a, bf16x16 b, f32x8 c) {
  // (neg_a, A, neg_b, B, c_mod, C, reuse_a, reuse_b)
  return __builtin_amdgcn_wmma_f32_16x16x32_bf16(false, a, false, b,
                                                 (short)0, c, false, false);
}

// Async copy of 16 bytes global -> LDS (ASYNCcnt path, no VGPR staging).
// LDS address comes from the first VGPR operand; generic __shared__ pointers carry the LDS
// byte offset in their low 32 bits.
__device__ __forceinline__ void async_copy_b128(uint32_t lds_off,
                                                const void* gptr) {
  asm volatile("global_load_async_to_lds_b128 %0, %1, off"
               :: "v"(lds_off), "v"((uint64_t)(uintptr_t)gptr)
               : "memory");
}
__device__ __forceinline__ void wait_async0() {
  asm volatile("s_wait_asynccnt 0x0" ::: "memory");
}

// Load one 16x32 bf16 fragment (A layout / mirrored B layout) from a
// row-major matrix: lanes 0-15 hold rows 0-15 with K 0-7 / 16-23,
// lanes 16-31 hold the same rows with K 8-15 / 24-31 (ISA 7.12.2).
__device__ __forceinline__ bf16x16 load_frag(const bf16* __restrict__ base,
                                             int ld, int lane) {
  const int r  = lane & 15;
  const int kl = (lane >> 4) << 3;
  const bf16* p = base + (size_t)r * ld + kl;
  bf16x8 lo = *reinterpret_cast<const bf16x8*>(p);        // global_load_b128
  bf16x8 hi = *reinterpret_cast<const bf16x8*>(p + 16);   // global_load_b128
  return __builtin_shufflevector(lo, hi, 0, 1, 2, 3, 4, 5, 6, 7,
                                 8, 9, 10, 11, 12, 13, 14, 15);
}

// ---------------------------------------------------------------- converts
__global__ __launch_bounds__(256) void cvt_f32_bf16_kernel(
    const float* __restrict__ in, bf16* __restrict__ out, int n4) {
  const int i = blockIdx.x * blockDim.x + threadIdx.x;
  if (i >= n4) return;
  const float4 v = reinterpret_cast<const float4*>(in)[i];
  bf16x4 o;
  o[0] = (bf16)v.x; o[1] = (bf16)v.y; o[2] = (bf16)v.z; o[3] = (bf16)v.w;
  reinterpret_cast<bf16x4*>(out)[i] = o;
}

// out[n][k] = (bf16) in[k][n]  (DD x DD), LDS-tiled for coalescing.
__global__ __launch_bounds__(256) void transpose_cvt_kernel(
    const float* __restrict__ in, bf16* __restrict__ out) {
  __shared__ float tile[32][33];
  const int tx = threadIdx.x & 31;
  const int ty = threadIdx.x >> 5;           // 0..7
  const int bx = blockIdx.x * 32;
  const int by = blockIdx.y * 32;
#pragma unroll
  for (int i = 0; i < 32; i += 8)
    tile[ty + i][tx] = in[(size_t)(by + ty + i) * DD + bx + tx];
  __syncthreads();
#pragma unroll
  for (int i = 0; i < 32; i += 8)
    out[(size_t)(bx + ty + i) * DD + by + tx] = (bf16)tile[tx][ty + i];
}

// -------------------------------------------------------------------- GEMM
// C[M,N] = A[M,K] (bf16, row-major) * BT[N,K]^T (bf16) + bias[N]
// mode 0: f32 out, row-major [M,N]           (final projection)
// mode 1: bf16 out, [b, h, t, hd]            (Q, K)
// mode 2: bf16 out, [b, h, hd, t]            (V transposed for PV WMMA)
// Block: 256 threads = 8 waves; tile 128M x 64N; weight tile double-buffered
// in LDS, filled with async global->LDS b128 copies (ASYNCcnt).
__global__ __launch_bounds__(256) void gemm_bf16_kernel(
    const bf16* __restrict__ A, const bf16* __restrict__ BT,
    const float* __restrict__ bias, void* __restrict__ out,
    int M, int N, int K, int mode) {
  __shared__ __align__(16) bf16 bs[2][64 * 32];   // 2 x 4 KB weight tiles
  const int tid  = threadIdx.x;
  const int lane = tid & 31;
  const int wave = tid >> 5;
  const int n0   = blockIdx.x * 64;
  const int m0   = blockIdx.y * 128 + wave * 16;

  f32x8 acc[4];
#pragma unroll
  for (int s = 0; s < 4; ++s) acc[s] = zero_f32x8();

  // Each of the 256 threads owns one b128 slot of the 64x32 tile.
  const int nl = tid >> 2;         // 0..63  (row of LDS tile)
  const int ko = (tid & 3) << 3;   // 0,8,16,24
  const uint32_t slot0 = (uint32_t)(uintptr_t)&bs[0][nl * 32 + ko];
  const bf16* gsrc = BT + (size_t)(n0 + nl) * K + ko;

  // Prologue: tile 0 -> buffer 0.
  async_copy_b128(slot0, gsrc);

  for (int k = 0; k < K; k += 32) {
    const int kb = (k >> 5) & 1;
    wait_async0();       // this wave's contribution to tile k has landed
    __syncthreads();     // everyone's contribution has landed

    // Overlap: start streaming tile k+32 into the other buffer.  The old
    // contents of that buffer were consumed (dscnt-waited) before every
    // wave passed the barrier above, so the overwrite cannot race.
    if (k + 32 < K)
      async_copy_b128(slot0 + (uint32_t)((kb ^ 1) * (64 * 32 * 2)),
                      gsrc + k + 32);

    const bf16x16 a = load_frag(A + (size_t)m0 * K + k, K, lane);

    const int rr = lane & 15;
    const int kl = (lane >> 4) << 3;
#pragma unroll
    for (int s = 0; s < 4; ++s) {
      const bf16* p = &bs[kb][(s * 16 + rr) * 32 + kl];
      bf16x8 lo = *reinterpret_cast<const bf16x8*>(p);       // ds_load_b128
      bf16x8 hi = *reinterpret_cast<const bf16x8*>(p + 16);  // ds_load_b128
      bf16x16 b = __builtin_shufflevector(lo, hi, 0, 1, 2, 3, 4, 5, 6, 7,
                                          8, 9, 10, 11, 12, 13, 14, 15);
      acc[s] = wmma_bf16(a, b, acc[s]);
    }
  }

  // C/D layout: lanes 0-15 -> rows m0+0..7, lanes 16-31 -> rows m0+8..15.
  const int mbase = m0 + ((lane >> 4) << 3);
#pragma unroll
  for (int s = 0; s < 4; ++s) {
    const int n  = n0 + s * 16 + (lane & 15);
    const float bv = bias[n];
    if (mode == 0) {
      float* op = (float*)out;
#pragma unroll
      for (int r = 0; r < 8; ++r)
        op[(size_t)(mbase + r) * N + n] = acc[s][r] + bv;
    } else if (mode == 1) {
      bf16* op = (bf16*)out;
      const int h = n >> 6, hd = n & 63;
#pragma unroll
      for (int r = 0; r < 8; ++r) {
        const int m = mbase + r, b = m >> 11, t = m & (TT - 1);
        op[(((size_t)b * HH + h) * TT + t) * HDD + hd] = (bf16)(acc[s][r] + bv);
      }
    } else {
      bf16* op = (bf16*)out;
      const int h = n >> 6, hd = n & 63;
#pragma unroll
      for (int r = 0; r < 8; ++r) {
        const int m = mbase + r, b = m >> 11, t = m & (TT - 1);
        op[(((size_t)b * HH + h) * HDD + hd) * TT + t] = (bf16)(acc[s][r] + bv);
      }
    }
  }
}

// --------------------------------------------------------- flash attention
// One wave handles one (b, h, 16-row q-tile); iterates 32-key tiles with
// online softmax.  S via 4 WMMAs, O += P*V via 4 WMMAs.  P is moved from
// the D-fragment layout to the A-fragment layout via a per-wave LDS bounce.
__global__ __launch_bounds__(128) void flash_attn_kernel(
    const bf16* __restrict__ q, const bf16* __restrict__ k,
    const bf16* __restrict__ vt, bf16* __restrict__ o) {
  __shared__ __align__(16) bf16 plds[4][16 * 32];   // 1 KB per wave
  const int lane = threadIdx.x & 31;
  const int wave = threadIdx.x >> 5;
  const int task = blockIdx.x * 4 + wave;           // 8192 tasks
  const int bh = task >> 7;                         // 0..63
  const int qt = task & 127;
  const int t0 = qt * 16;
  const int bb = bh >> 4, h = bh & 15;

  const bf16* qh  = q  + (size_t)bh * TT * HDD;     // [T][HD]
  const bf16* kh  = k  + (size_t)bh * TT * HDD;     // [T][HD]
  const bf16* vth = vt + (size_t)bh * HDD * TT;     // [HD][T]

  const bf16x16 qa0 = load_frag(qh + (size_t)t0 * HDD,      HDD, lane);
  const bf16x16 qa1 = load_frag(qh + (size_t)t0 * HDD + 32, HDD, lane);

  float mrow[8], lrow[8];
#pragma unroll
  for (int r = 0; r < 8; ++r) { mrow[r] = -3.0e38f; lrow[r] = 0.0f; }
  f32x8 oacc[4];
#pragma unroll
  for (int t = 0; t < 4; ++t) oacc[t] = zero_f32x8();

  const int rbase = (lane >> 4) << 3;   // row group of this half-wave
  const int c     = lane & 15;          // column held by this lane (C/D)
  const int kl    = (lane >> 4) << 3;   // K sub-offset (A/B frag)

  for (int j0 = 0; j0 <= t0; j0 += 32) {
    __builtin_prefetch(kh + (size_t)(j0 + 32) * HDD, 0, 1);

    f32x8 s0 = zero_f32x8(), s1 = zero_f32x8();
    {
      const bf16x16 kb0 = load_frag(kh + (size_t)j0 * HDD,      HDD, lane);
      const bf16x16 kb1 = load_frag(kh + (size_t)j0 * HDD + 32, HDD, lane);
      s0 = wmma_bf16(qa0, kb0, s0);
      s0 = wmma_bf16(qa1, kb1, s0);
      const bf16x16 kb2 = load_frag(kh + (size_t)(j0 + 16) * HDD,      HDD, lane);
      const bf16x16 kb3 = load_frag(kh + (size_t)(j0 + 16) * HDD + 32, HDD, lane);
      s1 = wmma_bf16(qa0, kb2, s1);
      s1 = wmma_bf16(qa1, kb3, s1);
    }
#pragma unroll
    for (int r = 0; r < 8; ++r) { s0[r] *= 0.125f; s1[r] *= 0.125f; }

    if (j0 + 31 > t0) {   // diagonal tile(s): causal mask key > query
#pragma unroll
      for (int r = 0; r < 8; ++r) {
        const int qi = t0 + rbase + r;
        if (j0 + c      > qi) s0[r] = -1.0e9f;
        if (j0 + 16 + c > qi) s1[r] = -1.0e9f;
      }
    }

    // Online softmax: rows live across 16-lane halves -> xor-shuffle reduce.
#pragma unroll
    for (int r = 0; r < 8; ++r) {
      float mx = fmaxf(s0[r], s1[r]);
      mx = fmaxf(mx, __shfl_xor(mx, 1, 32));
      mx = fmaxf(mx, __shfl_xor(mx, 2, 32));
      mx = fmaxf(mx, __shfl_xor(mx, 4, 32));
      mx = fmaxf(mx, __shfl_xor(mx, 8, 32));
      const float mn    = fmaxf(mrow[r], mx);
      const float alpha = __expf(mrow[r] - mn);
      mrow[r] = mn;
      const float p0 = __expf(s0[r] - mn);
      const float p1 = __expf(s1[r] - mn);
      s0[r] = p0; s1[r] = p1;
      float rs = p0 + p1;
      rs += __shfl_xor(rs, 1, 32);
      rs += __shfl_xor(rs, 2, 32);
      rs += __shfl_xor(rs, 4, 32);
      rs += __shfl_xor(rs, 8, 32);
      lrow[r] = lrow[r] * alpha + rs;
#pragma unroll
      for (int t = 0; t < 4; ++t) oacc[t][r] *= alpha;
    }

    // Transpose P (D layout -> A layout) through LDS; same-wave DS ops are
    // in order, s_wait_dscnt 0 guards the RAW.
#pragma unroll
    for (int r = 0; r < 8; ++r) {
      plds[wave][(rbase + r) * 32 + c]      = (bf16)s0[r];
      plds[wave][(rbase + r) * 32 + 16 + c] = (bf16)s1[r];
    }
    asm volatile("s_wait_dscnt 0x0" ::: "memory");

    bf16x16 pa;
    {
      const bf16* p = &plds[wave][(lane & 15) * 32 + kl];
      bf16x8 lo = *reinterpret_cast<const bf16x8*>(p);
      bf16x8 hi = *reinterpret_cast<const bf16x8*>(p + 16);
      pa = __builtin_shufflevector(lo, hi, 0, 1, 2, 3, 4, 5, 6, 7,
                                   8, 9, 10, 11, 12, 13, 14, 15);
    }

    // O += P * V using pre-transposed V (contiguous along keys).
#pragma unroll
    for (int t = 0; t < 4; ++t) {
      const bf16* vp = vth + (size_t)(t * 16 + c) * TT + j0 + kl;
      bf16x8 lo = *reinterpret_cast<const bf16x8*>(vp);
      bf16x8 hi = *reinterpret_cast<const bf16x8*>(vp + 16);
      bf16x16 vb = __builtin_shufflevector(lo, hi, 0, 1, 2, 3, 4, 5, 6, 7,
                                           8, 9, 10, 11, 12, 13, 14, 15);
      oacc[t] = wmma_bf16(pa, vb, oacc[t]);
    }
  }

  // Normalize and store O as [b, t, h*64 + hd] bf16 (GEMM-ready, row-major).
#pragma unroll
  for (int r = 0; r < 8; ++r) {
    const float linv = 1.0f / lrow[r];
    const size_t row = (size_t)(bb * TT + t0 + rbase + r);
#pragma unroll
    for (int t = 0; t < 4; ++t)
      o[row * DD + h * HDD + t * 16 + c] = (bf16)(oacc[t][r] * linv);
  }
}

// ------------------------------------------------------------------ launch
extern "C" void kernel_launch(void* const* d_in, const int* in_sizes, int n_in,
                              void* d_out, int out_size, void* d_ws,
                              size_t ws_size, hipStream_t stream) {
  (void)in_sizes; (void)n_in; (void)out_size; (void)ws_size;
  const float* x  = (const float*)d_in[0];
  const float* Wq = (const float*)d_in[1];
  const float* bq = (const float*)d_in[2];
  const float* Wk = (const float*)d_in[3];
  const float* bk = (const float*)d_in[4];
  const float* Wv = (const float*)d_in[5];
  const float* bv = (const float*)d_in[6];
  const float* Wp = (const float*)d_in[7];
  const float* bp = (const float*)d_in[8];
  float* out = (float*)d_out;

  // Workspace carve-up (bf16): 5 * 16MB activations + 4 * 2MB weights = 88MB.
  bf16* ws = (bf16*)d_ws;
  const size_t MD = (size_t)MTOT * DD;   // 8M elems
  const size_t WW = (size_t)DD * DD;     // 1M elems
  bf16* xb  = ws;
  bf16* wqT = xb  + MD;
  bf16* wkT = wqT + WW;
  bf16* wvT = wkT + WW;
  bf16* wpT = wvT + WW;
  bf16* qb  = wpT + WW;
  bf16* kb  = qb  + MD;
  bf16* vtb = kb  + MD;
  bf16* ao  = vtb + MD;

  cvt_f32_bf16_kernel<<<(int)(MD / 4 / 256), 256, 0, stream>>>(x, xb,
                                                               (int)(MD / 4));
  dim3 tg(DD / 32, DD / 32);
  transpose_cvt_kernel<<<tg, 256, 0, stream>>>(Wq, wqT);
  transpose_cvt_kernel<<<tg, 256, 0, stream>>>(Wk, wkT);
  transpose_cvt_kernel<<<tg, 256, 0, stream>>>(Wv, wvT);
  transpose_cvt_kernel<<<tg, 256, 0, stream>>>(Wp, wpT);

  dim3 gg(DD / 64, MTOT / 128);   // (16, 64)
  gemm_bf16_kernel<<<gg, 256, 0, stream>>>(xb, wqT, bq, qb,  MTOT, DD, DD, 1);
  gemm_bf16_kernel<<<gg, 256, 0, stream>>>(xb, wkT, bk, kb,  MTOT, DD, DD, 1);
  gemm_bf16_kernel<<<gg, 256, 0, stream>>>(xb, wvT, bv, vtb, MTOT, DD, DD, 2);

  flash_attn_kernel<<<(BB * HH * (TT / 16)) / 4, 128, 0, stream>>>(qb, kb, vtb,
                                                                   ao);

  gemm_bf16_kernel<<<gg, 256, 0, stream>>>(ao, wpT, bp, out, MTOT, DD, DD, 0);
}
